// LVQ_41042707480709
// MI455X (gfx1250) — matrix operand order repaired and checked by reading, not compiled, not run on hardware
//
#include <hip/hip_runtime.h>

#define NUM_CLASSES 1000
#define PPC 8
#define INPUT_DIM 64
#define BATCH 16384
#define NPROTO (NUM_CLASSES * PPC)

typedef __attribute__((ext_vector_type(16))) __bf16 v16bf;
typedef __attribute__((ext_vector_type(8)))  float  v8f;

// ---------------------------------------------------------------------------
// One-shot pack kernels: fp32 -> bf16 hi/lo split into WMMA fragment-major
// layout (exact per-lane register image), fused with squared-norm calc.
// Record: frag[(blk*2 + c)*32 + lane] = v16bf  (32B, lane-contiguous).
//
// Roles: prototypes are the A-matrix (M = proto), x is the B-matrix
// (N = sample), so the per-class PPC=8 reduction is register-local.
// ---------------------------------------------------------------------------

// Prototypes -> A-layout (16-bit A 16x32, this code's contiguous k order, not the ISA table): lane L: M = L&15, h = L>>4;
// element j: K = 16h + j, + 32c  (16 contiguous floats per chunk).
__global__ __launch_bounds__(32) void lvq_pack_protos(
    const float* __restrict__ p, v16bf* __restrict__ phi,
    v16bf* __restrict__ plo, float* __restrict__ pnorms) {
  const int lane = threadIdx.x & 31;
  const int m = lane & 15, h = lane >> 4;
  const int blk = blockIdx.x;  // 16-proto block (500 total)
  const float* row = p + (size_t)(blk * 16 + m) * INPUT_DIM;
  float s = 0.f;
#pragma unroll
  for (int c = 0; c < 2; ++c) {
    v16bf hi, lo;
    const float* r0 = row + c * 32 + h * 16;
    const float* r1 = row + c * 32 + h * 16 + 8;
#pragma unroll
    for (int j = 0; j < 8; ++j) {
      float f0 = r0[j]; s += f0 * f0;
      __bf16 h0 = (__bf16)f0;
      hi[j] = h0; lo[j] = (__bf16)(f0 - (float)h0);
      float f1 = r1[j]; s += f1 * f1;
      __bf16 h1 = (__bf16)f1;
      hi[8 + j] = h1; lo[8 + j] = (__bf16)(f1 - (float)h1);
    }
    phi[(blk * 2 + c) * 32 + lane] = hi;
    plo[(blk * 2 + c) * 32 + lane] = lo;
  }
  // lanes m and m+16 jointly hold all 64 elements of row m
  s += __shfl_xor(s, 16, 32);
  if (h == 0) pnorms[blk * 16 + m] = s;
}

// x -> B-layout (16-bit B 32x16, this code's contiguous k order, not the ISA table): lane L: N = L&15, h = L>>4;
// element j: K = 16h + j, + 32c  (16 contiguous floats per chunk).
__global__ __launch_bounds__(32) void lvq_pack_x(
    const float* __restrict__ x, v16bf* __restrict__ xhi,
    v16bf* __restrict__ xlo, float* __restrict__ xnorms) {
  const int lane = threadIdx.x & 31;
  const int m = lane & 15, h = lane >> 4;
  const int blk = blockIdx.x;  // 16-sample block (1024 total)
  const float* row = x + (size_t)(blk * 16 + m) * INPUT_DIM;
  float s = 0.f;
#pragma unroll
  for (int c = 0; c < 2; ++c) {
    v16bf hi, lo;
    const float* r0 = row + c * 32 + h * 16;
#pragma unroll
    for (int j = 0; j < 16; ++j) {
      float f = r0[j]; s += f * f;
      __bf16 hh = (__bf16)f;
      hi[j] = hh; lo[j] = (__bf16)(f - (float)hh);
    }
    xhi[(blk * 2 + c) * 32 + lane] = hi;
    xlo[(blk * 2 + c) * 32 + lane] = lo;
  }
  s += __shfl_xor(s, 16, 32);
  if (h == 0) xnorms[blk * 16 + m] = s;
}

// ---------------------------------------------------------------------------
// Main kernel: 4 waves/block; wave = 16 protos (1 A tile, 2 classes)
// x 64 samples (4 B tiles). acc[r] holds proto (8h + r) of this lane's
// class for sample (lane&15) -> class max is 7 register fmins, no shuffles.
// ---------------------------------------------------------------------------
__global__ __launch_bounds__(128) void lvq_wmma_kernel(
    const v16bf* __restrict__ phi, const v16bf* __restrict__ plo,
    const v16bf* __restrict__ xhi, const v16bf* __restrict__ xlo,
    const float* __restrict__ norms, float* __restrict__ out) {
  const int lane = threadIdx.x & 31;
  const int wave = threadIdx.x >> 5;
  const int m = lane & 15;
  const int h = lane >> 4;

  const int pblk = blockIdx.y * 4 + wave;  // proto tile 0..499
  const int cls  = pblk * 2 + h;           // this lane's output class

  // A fragments: prototypes (fixed for the whole wave)
  v16bf a_hi[2], a_lo[2];
#pragma unroll
  for (int c = 0; c < 2; ++c) {
    a_hi[c] = phi[(pblk * 2 + c) * 32 + lane];
    a_lo[c] = plo[(pblk * 2 + c) * 32 + lane];
  }

  // |p|^2 for the 8 protos this lane's acc registers cover (proto = 16*pblk+8h+r)
  const float* x2g = norms;
  const float* p2g = norms + BATCH;
  float pr2[8];
#pragma unroll
  for (int r = 0; r < 8; ++r) pr2[r] = p2g[pblk * 16 + h * 8 + r];

#pragma unroll
  for (int t = 0; t < 4; ++t) {
    const int nblk = blockIdx.x * 4 + t;  // sample tile 0..1023
    v16bf b_hi[2], b_lo[2];
#pragma unroll
    for (int c = 0; c < 2; ++c) {
      b_hi[c] = xhi[(nblk * 2 + c) * 32 + lane];
      b_lo[c] = xlo[(nblk * 2 + c) * 32 + lane];
    }

    // split-precision accumulate: hi*hi + lo*hi + hi*lo (~fp32 accuracy)
    v8f acc = {0.f, 0.f, 0.f, 0.f, 0.f, 0.f, 0.f, 0.f};
#pragma unroll
    for (int c = 0; c < 2; ++c) {
      acc = __builtin_amdgcn_wmma_f32_16x16x32_bf16(
          false, a_hi[c], false, b_hi[c], (short)0, acc, false, false);
      acc = __builtin_amdgcn_wmma_f32_16x16x32_bf16(
          false, a_lo[c], false, b_hi[c], (short)0, acc, false, false);
      acc = __builtin_amdgcn_wmma_f32_16x16x32_bf16(
          false, a_hi[c], false, b_lo[c], (short)0, acc, false, false);
    }

    // epilogue: min over the 8 protos of this class (register-local),
    // then one sqrt per lane. max(-sqrt(sq)) == -sqrt(min(sq)).
    float mn = __builtin_fmaf(acc[0], -2.0f, pr2[0]);
#pragma unroll
    for (int r = 1; r < 8; ++r)
      mn = fminf(mn, __builtin_fmaf(acc[r], -2.0f, pr2[r]));

    const int sample = nblk * 16 + m;
    float sq = fmaxf(x2g[sample] + mn, 0.0f);
    out[(size_t)sample * NUM_CLASSES + cls] = -__builtin_sqrtf(sq);
  }
}

extern "C" void kernel_launch(void* const* d_in, const int* in_sizes, int n_in,
                              void* d_out, int out_size, void* d_ws,
                              size_t ws_size, hipStream_t stream) {
  (void)in_sizes; (void)n_in; (void)out_size; (void)ws_size;
  const float* x = (const float*)d_in[0];
  const float* p = (const float*)d_in[1];
  float* out = (float*)d_out;  // [BATCH, NUM_CLASSES] fp32

  // Workspace layout (all offsets 32B-aligned):
  char* ws = (char*)d_ws;
  const size_t norms_bytes = (size_t)(BATCH + NPROTO) * sizeof(float); // 97536
  const size_t xfrag_bytes = (size_t)BATCH * INPUT_DIM * 2;            // 2 MB
  const size_t pfrag_bytes = (size_t)NPROTO * INPUT_DIM * 2;           // 1 MB
  float* norms = (float*)ws;
  v16bf* xhi = (v16bf*)(ws + norms_bytes);
  v16bf* xlo = (v16bf*)(ws + norms_bytes + xfrag_bytes);
  v16bf* phi = (v16bf*)(ws + norms_bytes + 2 * xfrag_bytes);
  v16bf* plo = (v16bf*)(ws + norms_bytes + 2 * xfrag_bytes + pfrag_bytes);

  lvq_pack_x<<<BATCH / 16, 32, 0, stream>>>(x, xhi, xlo, norms);
  lvq_pack_protos<<<NPROTO / 16, 32, 0, stream>>>(p, phi, plo, norms + BATCH);

  dim3 grid(BATCH / 64, NPROTO / 64);  // (256, 125); 4 proto-tiles per block
  lvq_wmma_kernel<<<grid, 128, 0, stream>>>(phi, plo, xhi, xlo, norms, out);
}
